// SwinUNETREncoder_29119878266905
// MI455X (gfx1250) — compile-verified, matrix operand order checked by reading, not run on hardware
//
#include <hip/hip_runtime.h>
#include <hip/hip_bf16.h>
#include <math.h>

typedef _Float16 v16h __attribute__((ext_vector_type(16)));
typedef float    v8f  __attribute__((ext_vector_type(8)));

union FragH {
    v16h v;
    _Float16 h[16];
    unsigned u[8];
};

// K-pair base for 16-bit A/B fragment VGPR v, lane-half `half` (ISA 7.12.2):
// lanes 0-15: V0..3 -> K 0..7, V4..7 -> K 16..23 ; lanes 16-31: +8
__device__ __forceinline__ int kmap(int v, int half) {
    return 2 * v + (v >= 4 ? 8 : 0) + 8 * half;
}

#define WMMA16(A, B, C) \
    __builtin_amdgcn_wmma_f32_16x16x32_f16(false, (A), false, (B), (short)0, (C), false, false)

// ---------------------------------------------------------------------------
// Convs
// ---------------------------------------------------------------------------
__global__ void adapter_conv(const float* __restrict__ in, const float* __restrict__ w,
                             const float* __restrict__ bias, float* __restrict__ out) {
    int idx = blockIdx.x * blockDim.x + threadIdx.x;
    if (idx >= 8 * 224 * 224) return;
    int x = idx % 224, y = (idx / 224) % 224, b = idx / (224 * 224);
    float acc = bias[0];
    for (int ci = 0; ci < 3; ++ci)
        for (int ky = 0; ky < 3; ++ky)
            for (int kx = 0; kx < 3; ++kx) {
                int iy = y + ky - 1, ix = x + kx - 1;
                if (iy >= 0 && iy < 224 && ix >= 0 && ix < 224)
                    acc += in[((size_t)(b * 3 + ci) * 224 + iy) * 224 + ix] *
                           w[(ci * 3 + ky) * 3 + kx];
            }
    out[idx] = acc;
}

__global__ void embed_conv(const float* __restrict__ a1, const float* __restrict__ w,
                           const float* __restrict__ bias, float* __restrict__ X) {
    int idx = blockIdx.x * blockDim.x + threadIdx.x;
    if (idx >= 8 * 112 * 112 * 48) return;
    int c = idx % 48;
    int t = idx / 48;
    int ox = t % 112, oy = (t / 112) % 112, b = t / (112 * 112);
    float acc = bias[c];
    for (int ky = 0; ky < 2; ++ky)
        for (int kx = 0; kx < 2; ++kx)
            acc += a1[((size_t)b * 224 + (2 * oy + ky)) * 224 + (2 * ox + kx)] *
                   w[c * 4 + ky * 2 + kx];
    X[idx] = acc;  // b,h,w,c
}

// ---------------------------------------------------------------------------
// LayerNorms
// ---------------------------------------------------------------------------
// LN fused with roll(-shift) + window partition: dest token t = win*49+pos
__global__ void ln_win(const float* __restrict__ X, float* __restrict__ XW,
                       const float* __restrict__ g, const float* __restrict__ bt,
                       int H, int W, int C, int shift, int Ntok) {
    int t = blockIdx.x * blockDim.x + threadIdx.x;
    if (t >= Ntok) return;
    int nwx = W / 7, nper = nwx * (H / 7);
    int pos = t % 49, win = t / 49;
    int b = win / nper, wi = win % nper;
    int ch = (wi / nwx) * 7 + pos / 7, cw = (wi % nwx) * 7 + pos % 7;
    int sh = (ch + shift) % H, sw = (cw + shift) % W;
    const float* row = X + ((size_t)(b * H + sh) * W + sw) * C;
    float m = 0.f;
    for (int c = 0; c < C; ++c) m += row[c];
    m /= C;
    float vv = 0.f;
    for (int c = 0; c < C; ++c) { float d = row[c] - m; vv += d * d; }
    vv /= C;
    float rs = rsqrtf(vv + 1e-5f);
    float* o = XW + (size_t)t * C;
    for (int c = 0; c < C; ++c) o[c] = (row[c] - m) * rs * g[c] + bt[c];
}

__global__ void ln_plain(const float* __restrict__ X, float* __restrict__ Y,
                         const float* __restrict__ g, const float* __restrict__ bt,
                         int C, int Ntok) {
    int t = blockIdx.x * blockDim.x + threadIdx.x;
    if (t >= Ntok) return;
    const float* row = X + (size_t)t * C;
    float m = 0.f;
    for (int c = 0; c < C; ++c) m += row[c];
    m /= C;
    float vv = 0.f;
    for (int c = 0; c < C; ++c) { float d = row[c] - m; vv += d * d; }
    vv /= C;
    float rs = rsqrtf(vv + 1e-5f);
    float* o = Y + (size_t)t * C;
    for (int c = 0; c < C; ++c) o[c] = (row[c] - m) * rs * g[c] + bt[c];
}

// ---------------------------------------------------------------------------
// GEMM: C[m][n] = sum_k A[m][k] * W[n][k] (+bias) with fused epilogue.
// A: MxK f32 row-major. W: NxK f32 row-major (torch Linear weight -> x @ W^T).
// Block = 8 waves, 32x64 block tile, 16x16 WMMA tile per wave, LDS-staged f16.
// K is always a multiple of 16 and rows 64B-aligned -> float4 fast-path loads.
// EPI: 0 store, 1 bias+GELU, 2 residual add, 3 window-reverse+roll residual scatter
// ---------------------------------------------------------------------------
template <int EPI>
__global__ __launch_bounds__(256) void gemm_f16w(const float* __restrict__ A,
                                                 const float* __restrict__ Wt,
                                                 const float* __restrict__ bias,
                                                 float* __restrict__ Out,
                                                 int M, int N, int K,
                                                 int H, int W, int shift) {
    __shared__ _Float16 As[32][36];
    __shared__ _Float16 Bs[64][36];
    int tid = threadIdx.x;
    int wave = tid >> 5, lane = tid & 31;
    int wm = wave >> 2, wn = wave & 3;
    int half = lane >> 4, l = lane & 15;
    int bm = blockIdx.x * 32, bn = blockIdx.y * 64;

    v8f acc = {0.f, 0.f, 0.f, 0.f, 0.f, 0.f, 0.f, 0.f};

    for (int k0 = 0; k0 < K; k0 += 32) {
        {   // A tile 32x32: 4 floats per thread (float4 fast path)
            int r = tid >> 3, c = (tid & 7) * 4;
            int gm = bm + r;
            const float* ap = A + (size_t)gm * K + k0 + c;
            if (gm < M && (k0 + c + 3) < K) {
                float4 vv = *(const float4*)ap;
                As[r][c + 0] = (_Float16)vv.x;
                As[r][c + 1] = (_Float16)vv.y;
                As[r][c + 2] = (_Float16)vv.z;
                As[r][c + 3] = (_Float16)vv.w;
            } else {
                #pragma unroll
                for (int i = 0; i < 4; ++i) {
                    int k = k0 + c + i;
                    As[r][c + i] = (_Float16)((gm < M && k < K) ? ap[i] : 0.f);
                }
            }
        }
        {   // B tile 64x32: 8 floats per thread (2x float4 fast path)
            int r = tid >> 2, c = (tid & 3) * 8;
            int gn = bn + r;
            const float* bp = Wt + (size_t)gn * K + k0 + c;
            if (gn < N && (k0 + c + 7) < K) {
                float4 v0 = *(const float4*)bp;
                float4 v1 = *(const float4*)(bp + 4);
                Bs[r][c + 0] = (_Float16)v0.x;
                Bs[r][c + 1] = (_Float16)v0.y;
                Bs[r][c + 2] = (_Float16)v0.z;
                Bs[r][c + 3] = (_Float16)v0.w;
                Bs[r][c + 4] = (_Float16)v1.x;
                Bs[r][c + 5] = (_Float16)v1.y;
                Bs[r][c + 6] = (_Float16)v1.z;
                Bs[r][c + 7] = (_Float16)v1.w;
            } else {
                #pragma unroll
                for (int i = 0; i < 8; ++i) {
                    int k = k0 + c + i;
                    Bs[r][c + i] = (_Float16)((gn < N && k < K) ? bp[i] : 0.f);
                }
            }
        }
        __syncthreads();

        FragH fa, fb;
        int ar = wm * 16 + l, br = wn * 16 + l;
        #pragma unroll
        for (int v = 0; v < 8; ++v) {
            int kb = kmap(v, half);
            fa.u[v] = *(const unsigned*)&As[ar][kb];
            fb.u[v] = *(const unsigned*)&Bs[br][kb];
        }
        acc = WMMA16(fa.v, fb.v, acc);
        __syncthreads();
    }

    int n = bn + wn * 16 + l;
    #pragma unroll
    for (int r = 0; r < 8; ++r) {
        int m = bm + wm * 16 + r + 8 * half;
        if (m < M && n < N) {
            float o = acc[r] + (bias ? bias[n] : 0.f);
            if (EPI == 0) {
                Out[(size_t)m * N + n] = o;
            } else if (EPI == 1) {
                Out[(size_t)m * N + n] = 0.5f * o * (1.f + erff(o * 0.70710678f));
            } else if (EPI == 2) {
                Out[(size_t)m * N + n] += o;
            } else {  // windowed token m -> canvas coords, roll(+shift), residual add
                int pos = m % 49, win = m / 49;
                int nwx = W / 7, nper = nwx * (H / 7);
                int b = win / nper, wi = win % nper;
                int ch = (wi / nwx) * 7 + pos / 7, cw = (wi % nwx) * 7 + pos % 7;
                int oh = (ch + shift) % H, ow = (cw + shift) % W;
                Out[((size_t)(b * H + oh) * W + ow) * N + n] += o;
            }
        }
    }
}

// ---------------------------------------------------------------------------
// Fused window attention: one block (4 waves) per (window, head). head_dim==16.
// S = (Q*0.25) K^T  (4x4 WMMA tiles, K padded 16->32) ; additive table
// BT[m][n] = rel-pos bias + shift mask (-1e30 at pads) precomputed in LDS so
// the S epilogue is branchless (s[r]==0 at padded rows/cols from zero-pad) ;
// row softmax ; O = P V (K=64 in 2 steps).
// ---------------------------------------------------------------------------
__device__ __forceinline__ int reg3(int c, int s) {
    return (c < s - 7) ? 0 : ((c < s - 3) ? 1 : 2);
}

__global__ __launch_bounds__(128) void win_attn(const float* __restrict__ qkv,
                                                float* __restrict__ att,
                                                const float* __restrict__ relb,
                                                int heads, int C, int H, int W, int shift) {
    int head = blockIdx.x % heads;
    int win = blockIdx.x / heads;
    int nwx = W / 7, nper = nwx * (H / 7);
    int wi = win % nper;
    int whi = wi / nwx, wwi = wi % nwx;
    int tid = threadIdx.x;
    int threeC = 3 * C;
    size_t winBase = (size_t)win * 49;

    __shared__ _Float16 Qb[64][36];
    __shared__ _Float16 Kb[64][36];
    __shared__ _Float16 Vb[64][16];
    __shared__ float    Sf[64][65];
    __shared__ _Float16 Pb[64][68];
    __shared__ float    BiasT[169];
    __shared__ int      LabT[49];
    __shared__ float    BT[64][65];

    // preload per-head rel-pos bias column and (if shifted) region labels
    for (int e = tid; e < 169; e += 128) BiasT[e] = relb[e * heads + head];
    if (shift)
        for (int e = tid; e < 49; e += 128)
            LabT[e] = reg3(whi * 7 + e / 7, H) * 3 + reg3(wwi * 7 + e % 7, W);

    // stage Q(scaled)/K/V with float4 loads; zero-pad rows 49..63, K-cols 16..31
    for (int e = tid; e < 256; e += 128) {
        int r = e >> 2, d4 = (e & 3) * 4;
        float4 q = {0.f, 0.f, 0.f, 0.f}, k = q, v = q;
        if (r < 49) {
            const float* base = qkv + (winBase + r) * threeC + head * 16 + d4;
            q = *(const float4*)base;
            k = *(const float4*)(base + C);
            v = *(const float4*)(base + 2 * C);
        }
        Qb[r][d4 + 0] = (_Float16)(q.x * 0.25f);
        Qb[r][d4 + 1] = (_Float16)(q.y * 0.25f);
        Qb[r][d4 + 2] = (_Float16)(q.z * 0.25f);
        Qb[r][d4 + 3] = (_Float16)(q.w * 0.25f);
        Kb[r][d4 + 0] = (_Float16)k.x;
        Kb[r][d4 + 1] = (_Float16)k.y;
        Kb[r][d4 + 2] = (_Float16)k.z;
        Kb[r][d4 + 3] = (_Float16)k.w;
        Vb[r][d4 + 0] = (_Float16)v.x;
        Vb[r][d4 + 1] = (_Float16)v.y;
        Vb[r][d4 + 2] = (_Float16)v.z;
        Vb[r][d4 + 3] = (_Float16)v.w;
        Qb[r][d4 + 16] = (_Float16)0.f;
        Kb[r][d4 + 16] = (_Float16)0.f;
        Qb[r][d4 + 17] = (_Float16)0.f;
        Kb[r][d4 + 17] = (_Float16)0.f;
        Qb[r][d4 + 18] = (_Float16)0.f;
        Kb[r][d4 + 18] = (_Float16)0.f;
        Qb[r][d4 + 19] = (_Float16)0.f;
        Kb[r][d4 + 19] = (_Float16)0.f;
    }
    __syncthreads();

    // build branchless additive table BT (bias + mask, -1e30 at pads)
    for (int e = tid; e < 4096; e += 128) {
        int m = e >> 6, n = e & 63;
        float t = -1e30f;
        if (m < 49 && n < 49) {
            int ridx = (m / 7 - n / 7 + 6) * 13 + (m % 7 - n % 7 + 6);
            t = BiasT[ridx];
            if (shift && (LabT[m] != LabT[n])) t -= 100.f;
        }
        BT[m][n] = t;
    }
    __syncthreads();

    int wave = tid >> 5, lane = tid & 31, half = lane >> 4, l = lane & 15;

    // S = Q K^T ; wave handles m-tile = wave, loops n-tiles ; branchless epilogue
    {
        FragH fa;
        int ar = wave * 16 + l;
        #pragma unroll
        for (int v = 0; v < 8; ++v) fa.u[v] = *(const unsigned*)&Qb[ar][kmap(v, half)];
        for (int j = 0; j < 4; ++j) {
            FragH fb;
            int br = j * 16 + l;
            #pragma unroll
            for (int v = 0; v < 8; ++v) fb.u[v] = *(const unsigned*)&Kb[br][kmap(v, half)];
            v8f s = {0.f, 0.f, 0.f, 0.f, 0.f, 0.f, 0.f, 0.f};
            s = WMMA16(fa.v, fb.v, s);
            int n = j * 16 + l;
            #pragma unroll
            for (int r = 0; r < 8; ++r) {
                int m = wave * 16 + r + 8 * half;
                Sf[m][n] = s[r] + BT[m][n];
            }
        }
    }
    __syncthreads();

    // per-row softmax over 49 keys -> Pb (f16), zero-pad to 64
    if (tid < 64) {
        int r = tid;
        if (r < 49) {
            float mx = -1e30f;
            for (int n = 0; n < 49; ++n) mx = fmaxf(mx, Sf[r][n]);
            float sum = 0.f;
            for (int n = 0; n < 49; ++n) {
                float e = expf(Sf[r][n] - mx);
                sum += e;
                Pb[r][n] = (_Float16)e;
            }
            float inv = 1.f / sum;
            for (int n = 0; n < 49; ++n) Pb[r][n] = (_Float16)((float)Pb[r][n] * inv);
            for (int n = 49; n < 64; ++n) Pb[r][n] = (_Float16)0.f;
        } else {
            for (int n = 0; n < 64; ++n) Pb[r][n] = (_Float16)0.f;
        }
    }
    __syncthreads();

    // O = P V ; K = 64 in two WMMA steps
    {
        v8f o = {0.f, 0.f, 0.f, 0.f, 0.f, 0.f, 0.f, 0.f};
        int ar = wave * 16 + l;
        #pragma unroll
        for (int k0 = 0; k0 < 64; k0 += 32) {
            FragH fa, fb;
            #pragma unroll
            for (int v = 0; v < 8; ++v) {
                int kb = kmap(v, half);
                fa.u[v] = *(const unsigned*)&Pb[ar][k0 + kb];
                fb.h[2 * v] = Vb[k0 + kb][l];
                fb.h[2 * v + 1] = Vb[k0 + kb + 1][l];
            }
            o = WMMA16(fa.v, fb.v, o);
        }
        #pragma unroll
        for (int r = 0; r < 8; ++r) {
            int m = wave * 16 + r + 8 * half;
            if (m < 49)
                att[(winBase + m) * C + head * 16 + l] = o[r];
        }
    }
}

// ---------------------------------------------------------------------------
// Patch merge gather (2x2 concat, MONAI order) and CHW output transpose
// ---------------------------------------------------------------------------
__global__ void merge_gather(const float* __restrict__ X, float* __restrict__ out,
                             int H, int W, int C, int total) {
    int idx = blockIdx.x * blockDim.x + threadIdx.x;
    if (idx >= total) return;
    int C4 = 4 * C;
    int cc = idx % C4;
    int t = idx / C4;
    int W2 = W / 2, H2 = H / 2;
    int w2 = t % W2, h2 = (t / W2) % H2, b = t / (W2 * H2);
    int q = cc / C, c = cc % C;
    int dh = q & 1, dw = q >> 1;  // (0,0),(1,0),(0,1),(1,1)
    out[idx] = X[((size_t)(b * H + 2 * h2 + dh) * W + (2 * w2 + dw)) * C + c];
}

__global__ void to_chw(const float* __restrict__ X, float* __restrict__ out,
                       int H, int W, int C, int total) {
    int idx = blockIdx.x * blockDim.x + threadIdx.x;
    if (idx >= total) return;
    int c = idx % C;
    int t = idx / C;
    int wq = t % W, hq = (t / W) % H, b = t / (W * H);
    out[((size_t)(b * C + c) * H + hq) * W + wq] = X[idx];
}

// ---------------------------------------------------------------------------
// Orchestration
// ---------------------------------------------------------------------------
extern "C" void kernel_launch(void* const* d_in, const int* in_sizes, int n_in,
                              void* d_out, int out_size, void* d_ws, size_t ws_size,
                              hipStream_t stream) {
    (void)in_sizes; (void)n_in; (void)out_size; (void)ws_size;
    int ii = 0;
    auto F = [&](void) { return (const float*)d_in[ii++]; };

    const float* x_in = F();
    const float* aw = F(); const float* ab = F();
    const float* ew = F(); const float* eb = F();

    struct Blk {
        const float *n1g, *n1b, *qkvw, *qkvb, *relb, *projw, *projb;
        const float *n2g, *n2b, *f1w, *f1b, *f2w, *f2b;
    };
    struct Stg { Blk blk[2]; const float *dsg, *dsb, *dsr; } st[4];
    for (int s = 0; s < 4; ++s) {
        for (int b = 0; b < 2; ++b) {
            Blk& k = st[s].blk[b];
            k.n1g = F(); k.n1b = F();
            k.qkvw = F(); k.qkvb = F();
            k.relb = F();
            k.projw = F(); k.projb = F();
            k.n2g = F(); k.n2b = F();
            k.f1w = F(); k.f1b = F();
            k.f2w = F(); k.f2b = F();
        }
        st[s].dsg = F(); st[s].dsb = F(); st[s].dsr = F();
    }

    float* ws = (float*)d_ws;
    float* X   = ws;                    // cap 4816896 (stage-1 b,h,w,c)
    float* X2  = X + 4816896;           // cap 2408448 (max merged output)
    float* XW  = X2 + 2408448;          // cap 4816896 (LN/windowed tokens)
    float* QKV = XW + 4816896;          // cap 14450688 (stage-1 Ntok*3C)
    float* ATT = QKV + 14450688;        // cap 4816896
    float* HB  = ATT + 4816896;         // cap 19267584 (stage-1 Ntok*4C)
    float* A1  = HB + 19267584;         // adapter output, 401408
    float* fo  = (float*)d_out;

    {   int n = 8 * 224 * 224;
        adapter_conv<<<(n + 255) / 256, 256, 0, stream>>>(x_in, aw, ab, A1); }
    {   int n = 8 * 112 * 112 * 48;
        embed_conv<<<(n + 255) / 256, 256, 0, stream>>>(A1, ew, eb, X); }

    size_t foff = 0;
    for (int s = 0; s < 4; ++s) {
        int Hc = 112 >> s, Wc = 112 >> s, C = 48 << s, heads = 3 << s;
        int Ntok = 8 * Hc * Wc;
        int nWin = Ntok / 49;

        for (int bi = 0; bi < 2; ++bi) {
            const Blk& p = st[s].blk[bi];
            int shift = bi ? 3 : 0;

            ln_win<<<(Ntok + 255) / 256, 256, 0, stream>>>(X, XW, p.n1g, p.n1b,
                                                           Hc, Wc, C, shift, Ntok);
            dim3 g1((Ntok + 31) / 32, (3 * C + 63) / 64);
            gemm_f16w<0><<<g1, 256, 0, stream>>>(XW, p.qkvw, p.qkvb, QKV,
                                                 Ntok, 3 * C, C, Hc, Wc, shift);
            win_attn<<<nWin * heads, 128, 0, stream>>>(QKV, ATT, p.relb,
                                                       heads, C, Hc, Wc, shift);
            dim3 g2((Ntok + 31) / 32, (C + 63) / 64);
            gemm_f16w<3><<<g2, 256, 0, stream>>>(ATT, p.projw, p.projb, X,
                                                 Ntok, C, C, Hc, Wc, shift);
            ln_plain<<<(Ntok + 255) / 256, 256, 0, stream>>>(X, XW, p.n2g, p.n2b, C, Ntok);
            dim3 g3((Ntok + 31) / 32, (4 * C + 63) / 64);
            gemm_f16w<1><<<g3, 256, 0, stream>>>(XW, p.f1w, p.f1b, HB,
                                                 Ntok, 4 * C, C, Hc, Wc, 0);
            dim3 g4((Ntok + 31) / 32, (C + 63) / 64);
            gemm_f16w<2><<<g4, 256, 0, stream>>>(HB, p.f2w, p.f2b, X,
                                                 Ntok, C, 4 * C, Hc, Wc, 0);
        }

        // patch merge
        int H2 = Hc / 2, W2 = Wc / 2, C2 = 2 * C, Nt2 = 8 * H2 * W2;
        {   int n = Nt2 * 4 * C;
            merge_gather<<<(n + 255) / 256, 256, 0, stream>>>(X, HB, Hc, Wc, C, n); }
        ln_plain<<<(Nt2 + 255) / 256, 256, 0, stream>>>(HB, XW, st[s].dsg, st[s].dsb,
                                                        4 * C, Nt2);
        dim3 g5((Nt2 + 31) / 32, (C2 + 63) / 64);
        gemm_f16w<0><<<g5, 256, 0, stream>>>(XW, st[s].dsr, nullptr, X2,
                                             Nt2, C2, 4 * C, Hc, Wc, 0);
        {   int n = Nt2 * C2;
            to_chw<<<(n + 255) / 256, 256, 0, stream>>>(X2, fo + foff, H2, W2, C2, n); }
        foff += (size_t)Nt2 * C2;

        float* t = X; X = X2; X2 = t;  // merged output becomes next-stage input
    }
}
